// GraphSAGE_22376779612323
// MI455X (gfx1250) — matrix operand order repaired and checked by reading, not compiled, not run on hardware
//
#include <hip/hip_runtime.h>
#include <hip/hip_bf16.h>
#include <math.h>

typedef __attribute__((ext_vector_type(16))) __bf16 v16bf;
typedef __attribute__((ext_vector_type(8)))  float  v8f;

// ---------------------------------------------------------------- utilities
__global__ void zero_f32_kernel(float* __restrict__ p, long long n) {
  long long i = (long long)blockIdx.x * blockDim.x + threadIdx.x;
  if (i < n) p[i] = 0.0f;
}

__global__ void degree_kernel(const long long* __restrict__ dst,
                              float* __restrict__ deg, int e) {
  int i = blockIdx.x * blockDim.x + threadIdx.x;
  if (i < e) atomicAdd(&deg[(int)dst[i]], 1.0f);
}

__global__ void invdeg_kernel(float* __restrict__ deg, int n) {
  int i = blockIdx.x * blockDim.x + threadIdx.x;
  if (i < n) deg[i] = 1.0f / fmaxf(deg[i], 1.0f);
}

// one thread per (edge, feature): lanes cover consecutive features of one edge
// -> src/dst index loads are wave-uniform broadcasts, the x[src] gather and the
// agg[dst] atomic burst are coalesced 128B segments (all L2-resident: x is 25.6MB)
__global__ void scatter_kernel(const float* __restrict__ x,
                               const long long* __restrict__ src,
                               const long long* __restrict__ dst,
                               float* __restrict__ agg, long long total) {
  long long t = (long long)blockIdx.x * blockDim.x + threadIdx.x;
  if (t >= total) return;
  int e = (int)(t >> 6);
  int f = (int)(t & 63);
  int s = (int)src[e];
  int d = (int)dst[e];
  atomicAdd(&agg[d * 64 + f], x[s * 64 + f]);
}

// ------------------------------------------------ fused SAGE GEMM via WMMA
// out[N,HO] = relu( [agg*invdeg | xin] (N x 128, bf16) @ [Wl;Wr] (128 x HO, bf16) + b )
// 8 waves / block, each wave owns a 16-row tile; K = 128 = 4 WMMA k-steps.
template <int HO_PAD, int HO>
__global__ __launch_bounds__(256) void sage_gemm_kernel(
    const float* __restrict__ agg, const float* __restrict__ invdeg,
    const float* __restrict__ xin,
    const float* __restrict__ Wl, const float* __restrict__ Wr,
    const float* __restrict__ bias,
    float* __restrict__ out, int n) {
  __shared__ __align__(16) __bf16 Wt[HO_PAD][128];     // weights transposed [col][k]
  __shared__ float bsh[HO_PAD];
  __shared__ __align__(16) __bf16 Atile[8][16][128];   // per-wave A tiles

  const int tid = threadIdx.x;

  // stage concat weights [Wl;Wr] transposed to [col][k] as bf16
  for (int idx = tid; idx < HO_PAD * 128; idx += 256) {
    int col = idx >> 7;
    int k   = idx & 127;
    float w = 0.0f;
    if (col < HO) w = (k < 64) ? Wl[k * HO + col] : Wr[(k - 64) * HO + col];
    Wt[col][k] = (__bf16)w;
  }
  if (tid < HO_PAD) bsh[tid] = (tid < HO) ? bias[tid] : 0.0f;
  __syncthreads();

  const int wave    = tid >> 5;
  const int lane    = tid & 31;
  const int rowBase = blockIdx.x * 128 + wave * 16;

  // stage A tile: rows = nodes, K 0..63 = mean-aggregated neighbors, K 64..127 = self
  {
    int r    = lane & 15;
    int half = lane >> 4;
    int row  = rowBase + r;
    __bf16* dp = &Atile[wave][r][half * 64];
    if (row < n) {
      float scale = half ? 1.0f : invdeg[row];
      const float4* sp = (const float4*)((half ? xin : agg) + row * 64);
#pragma unroll
      for (int j = 0; j < 16; ++j) {
        float4 v = sp[j];
        dp[4 * j + 0] = (__bf16)(v.x * scale);
        dp[4 * j + 1] = (__bf16)(v.y * scale);
        dp[4 * j + 2] = (__bf16)(v.z * scale);
        dp[4 * j + 3] = (__bf16)(v.w * scale);
      }
    } else {
#pragma unroll
      for (int j = 0; j < 64; ++j) dp[j] = (__bf16)0.0f;
    }
  }
  __syncthreads();

  union Frag { v16bf v; uint4 q[2]; };

  // ISA 16-bit A 16x32 layout: lanes 0-15 hold K {0..7,16..23}, lanes 16-31 {8..15,24..31}
  const int ra = lane & 15;
  const int ao = (lane >= 16) ? 8 : 0;
  // B 32x16: lanes 0-15 col=lane K {0..7,16..23}, lanes 16-31 col=lane-16 K {8..15,24..31}
  const int cb = lane & 15;
  const int ko = (lane >= 16) ? 8 : 0;
  const int rofs = (lane >= 16) ? 8 : 0;
  const bool fullTile = (rowBase + 16 <= n);

#pragma unroll
  for (int c = 0; c < HO_PAD / 16; ++c) {
    v8f acc = {};
#pragma unroll
    for (int kk = 0; kk < 4; ++kk) {
      Frag a, b;
      const __bf16* ap = &Atile[wave][ra][kk * 32 + ao];
      a.q[0] = *(const uint4*)(ap);
      a.q[1] = *(const uint4*)(ap + 16);
      const __bf16* bp = &Wt[c * 16 + cb][kk * 32 + ko];
      b.q[0] = *(const uint4*)(bp);
      b.q[1] = *(const uint4*)(bp + 16);
      acc = __builtin_amdgcn_wmma_f32_16x16x32_bf16(
          false, a.v, false, b.v, (short)0, acc, false, false);
    }
    // D layout: VGPR i -> row = i + 8*(lane>=16), col = lane&15
    const int col = c * 16 + (lane & 15);
    if (HO == HO_PAD || col < HO) {        // compile-time true for HO=64
      const float bv = bsh[col];
      float* op = out + (rowBase + rofs) * HO + col;   // 32-bit addressing
      if (fullTile) {
#pragma unroll
        for (int i = 0; i < 8; ++i)
          op[i * HO] = fmaxf(acc[i] + bv, 0.0f);
      } else {
#pragma unroll
        for (int i = 0; i < 8; ++i) {
          if (rowBase + rofs + i < n)
            op[i * HO] = fmaxf(acc[i] + bv, 0.0f);
        }
      }
    }
  }
}

// ------------------------------------------------------------- log-softmax
__global__ void log_softmax_kernel(float* __restrict__ o, int n, int c) {
  int i = blockIdx.x * blockDim.x + threadIdx.x;
  if (i >= n) return;
  float* row = o + i * c;
  float m = -1e30f;
  for (int j = 0; j < c; ++j) m = fmaxf(m, row[j]);
  float s = 0.0f;
  for (int j = 0; j < c; ++j) s += __expf(row[j] - m);
  float lse = m + __logf(s);
  for (int j = 0; j < c; ++j) row[j] -= lse;
}

// ------------------------------------------------------------------ driver
extern "C" void kernel_launch(void* const* d_in, const int* in_sizes, int n_in,
                              void* d_out, int out_size, void* d_ws,
                              size_t ws_size, hipStream_t stream) {
  (void)n_in; (void)out_size; (void)ws_size;
  const float*     x   = (const float*)d_in[0];
  const long long* ei  = (const long long*)d_in[1];   // int64 edge_index [2,E]
  const float*     Wl1 = (const float*)d_in[2];
  const float*     Wr1 = (const float*)d_in[3];
  const float*     b1  = (const float*)d_in[4];
  const float*     Wl2 = (const float*)d_in[5];
  const float*     Wr2 = (const float*)d_in[6];
  const float*     b2  = (const float*)d_in[7];
  const float*     Wl3 = (const float*)d_in[8];
  const float*     Wr3 = (const float*)d_in[9];
  const float*     b3  = (const float*)d_in[10];

  const int F = 64;
  const int N = in_sizes[0] / F;
  const int E = in_sizes[1] / 2;
  const int C = in_sizes[10];
  const long long* src  = ei;
  const long long* dstv = ei + E;

  // workspace carve-out
  char* ws = (char*)d_ws;
  size_t off = 0;
  auto carve = [&](size_t bytes) -> void* {
    void* p = ws + off;
    off = (off + bytes + 255) & ~(size_t)255;
    return p;
  };
  float* invdeg = (float*)carve((size_t)N * 4);
  float* agg    = (float*)carve((size_t)N * F * 4);
  float* hA     = (float*)carve((size_t)N * F * 4);
  float* hB     = (float*)carve((size_t)N * F * 4);

  const int T = 256;
  const long long scat = (long long)E * F;
  const int scatBlocks = (int)((scat + T - 1) / T);
  const long long aggN = (long long)N * F;
  const int aggBlocks  = (int)((aggN + T - 1) / T);
  const int gemmBlocks = (N + 127) / 128;

  // degrees (reused across all layers)
  zero_f32_kernel<<<(N + T - 1) / T, T, 0, stream>>>(invdeg, N);
  degree_kernel<<<(E + T - 1) / T, T, 0, stream>>>(dstv, invdeg, E);
  invdeg_kernel<<<(N + T - 1) / T, T, 0, stream>>>(invdeg, N);

  // layer 1: x -> hA
  zero_f32_kernel<<<aggBlocks, T, 0, stream>>>(agg, aggN);
  scatter_kernel<<<scatBlocks, T, 0, stream>>>(x, src, dstv, agg, scat);
  sage_gemm_kernel<64, 64><<<gemmBlocks, T, 0, stream>>>(agg, invdeg, x, Wl1,
                                                         Wr1, b1, hA, N);
  // layer 2: hA -> hB
  zero_f32_kernel<<<aggBlocks, T, 0, stream>>>(agg, aggN);
  scatter_kernel<<<scatBlocks, T, 0, stream>>>(hA, src, dstv, agg, scat);
  sage_gemm_kernel<64, 64><<<gemmBlocks, T, 0, stream>>>(agg, invdeg, hA, Wl2,
                                                         Wr2, b2, hB, N);
  // layer 3: hB -> d_out (C=40 padded to 48 tiles), then log-softmax in place
  zero_f32_kernel<<<aggBlocks, T, 0, stream>>>(agg, aggN);
  scatter_kernel<<<scatBlocks, T, 0, stream>>>(hB, src, dstv, agg, scat);
  sage_gemm_kernel<48, 40><<<gemmBlocks, T, 0, stream>>>(agg, invdeg, hB, Wl3,
                                                         Wr3, b3,
                                                         (float*)d_out, N);
  log_softmax_kernel<<<(N + T - 1) / T, T, 0, stream>>>((float*)d_out, N, C);
}
